// SimpleSNNModel1D_79216376807540
// MI455X (gfx1250) — compile-verified, matrix operand order checked by reading, not run on hardware
//
#include <hip/hip_runtime.h>
#include <hip/hip_bf16.h>

typedef __attribute__((ext_vector_type(16))) _Float16 v16h;
typedef __attribute__((ext_vector_type(8)))  float    v8f;

// Scheduling-pipeline control (mask: 0x100 = DS read, 0x8 = MFMA/WMMA).
#if defined(__has_builtin)
#  if __has_builtin(__builtin_amdgcn_sched_group_barrier)
#    define SGB(mask, size, id) __builtin_amdgcn_sched_group_barrier(mask, size, id)
#  endif
#endif
#ifndef SGB
#  define SGB(mask, size, id)
#endif

// ---------------------------------------------------------------------------
// Workspace layout (bytes):
//   [0      , 131072)  w2h      : fp16 copy of w2, row-major [g][h] (256x256)
//   [131072 , 135168)  we1pack  : float4[256]  (w1*inv1 x3, fused bias1)
//   [135168 , 137216)  invbias2 : float2[256]  (inv2, fused bias2)
//   [137216 , 202752)  feat     : float[64*256] spike-sum accumulator
// ---------------------------------------------------------------------------
#define WS_W2H      0
#define WS_WE1      131072
#define WS_INVB2    135168
#define WS_FEAT     137216

// Prep: fold BN scales into weights/biases, convert w2 to fp16, zero feat.
__global__ __launch_bounds__(256) void snn_prep(
    const float* __restrict__ w1, const float* __restrict__ b1,
    const float* __restrict__ g1, const float* __restrict__ be1,
    const float* __restrict__ m1, const float* __restrict__ v1,
    const float* __restrict__ w2, const float* __restrict__ b2,
    const float* __restrict__ g2, const float* __restrict__ be2,
    const float* __restrict__ m2, const float* __restrict__ v2,
    _Float16* __restrict__ w2h, float4* __restrict__ we1pack,
    float2* __restrict__ invbias2, float* __restrict__ feat)
{
    int tid = blockIdx.x * 256 + threadIdx.x;
    if (tid < 65536) w2h[tid] = (_Float16)w2[tid];
    if (tid < 16384) feat[tid] = 0.0f;
    if (tid < 256) {
        float inv1  = g1[tid] * rsqrtf(v1[tid] + 1e-5f);
        float bias1 = b1[tid] * inv1 + be1[tid] - m1[tid] * inv1;
        we1pack[tid] = make_float4(w1[tid*3+0]*inv1, w1[tid*3+1]*inv1,
                                   w1[tid*3+2]*inv1, bias1);
        float inv2  = g2[tid] * rsqrtf(v2[tid] + 1e-5f);
        float bias2 = b2[tid] * inv2 + be2[tid] - m2[tid] * inv2;
        invbias2[tid] = make_float2(inv2, bias2);
    }
}

// Fused: conv1(K=3)+BN1+LIF1 -> LDS spikes -> WMMA GEMM(K=256)+BN2+LIF2 -> mean.
// Block = (g-tile of 64, batch b). 8 waves: wave w -> tile (m = w>>1, n = w&1).
__global__ __launch_bounds__(256) void snn_main(
    const float*   __restrict__ x,        // (B=64, T=64, C=3, V=32)
    const _Float16* __restrict__ w2h,     // [256][256] fp16
    const float4*  __restrict__ we1pack,  // [256]
    const float2*  __restrict__ invbias2, // [256]
    float*         __restrict__ feat)     // [64][256] spike sums
{
    __shared__ _Float16 s1LDS[32 * 256];  // [v][h] spike tile, 16 KB
    __shared__ float    xs[96];           // x_t staged: [c][v]

    const int tid   = threadIdx.x;
    const int b     = blockIdx.y;
    const int gbase = blockIdx.x * 64;
    const int lane  = tid & 31;
    const int wave  = tid >> 5;
    const int m     = wave >> 1;          // 0..3 : M (g) tile
    const int n     = wave & 1;           // 0..1 : N (v) tile

    // --- A fragments (W2 fp16) pinned in VGPRs for the whole kernel -------
    // A 16x32 fp16 layout: lane l holds row M = l&15; K half-swap per lane half.
    const int kb = (lane >> 4) * 8;
    const int gA = gbase + m * 16 + (lane & 15);
    v16h af[8];
#pragma unroll
    for (int kk = 0; kk < 8; ++kk) {
        const _Float16* p = w2h + gA * 256 + kk * 32 + kb;
        *(float4*)&af[kk]       = *(const float4*)p;         // K = kb..kb+7
        *((float4*)&af[kk] + 1) = *(const float4*)(p + 16);  // K = kb+16..kb+23
    }

    // --- per-lane BN2 constants for the 8 accumulator rows ----------------
    float inv2r[8], bias2r[8];
#pragma unroll
    for (int r = 0; r < 8; ++r) {
        int gr = gbase + m * 16 + r + ((lane >> 4) << 3);   // D: M = r (+8 hi half)
        float2 ib = invbias2[gr];
        inv2r[r] = ib.x; bias2r[r] = ib.y;
    }

    // --- LIF1: thread owns h = (tid&31)*8 .. +7  x  v = (tid>>5)*4 .. +3 --
    // Weights held in registers (8 float4), loaded once.
    const int hgrp = tid & 31;            // h block of 8
    const int vgrp = tid >> 5;            // v block of 4
    float4 wreg[8];
#pragma unroll
    for (int j = 0; j < 8; ++j) wreg[j] = we1pack[hgrp * 8 + j];

    float st1[32];                        // [v 0..3][h 0..7]
#pragma unroll
    for (int i = 0; i < 32; ++i) st1[i] = 0.0f;

    float st2[8], ssum[8];
#pragma unroll
    for (int r = 0; r < 8; ++r) { st2[r] = 0.0f; ssum[r] = 0.0f; }

    const float* xb = x + (size_t)b * 64 * 96;   // b * T*C*V
    const int vB = n * 16 + (lane & 15);         // B frag: N = lane&15

    for (int t = 0; t < 64; ++t) {
        if (tid < 96) xs[tid] = xb[t * 96 + tid];
        __syncthreads();  // xs ready; prior iter's s1LDS reads complete

        // ---- layer1 (K=3) + BN1 + LIF1 -> spikes into LDS ----------------
        float4 xc0 = *(const float4*)&xs[ 0 + vgrp * 4];
        float4 xc1 = *(const float4*)&xs[32 + vgrp * 4];
        float4 xc2 = *(const float4*)&xs[64 + vgrp * 4];
        const float xv0[4] = {xc0.x, xc0.y, xc0.z, xc0.w};
        const float xv1[4] = {xc1.x, xc1.y, xc1.z, xc1.w};
        const float xv2[4] = {xc2.x, xc2.y, xc2.z, xc2.w};
#pragma unroll
        for (int i = 0; i < 4; ++i) {                 // v = vgrp*4 + i
            union { float4 f4; _Float16 h[8]; } ch;
#pragma unroll
            for (int j = 0; j < 8; ++j) {             // h = hgrp*8 + j
                int s = i * 8 + j;
                float4 w = wreg[j];
                float a  = fmaf(w.x, xv0[i], fmaf(w.y, xv1[i], fmaf(w.z, xv2[i], w.w)));
                float Ht = fmaf(a - st1[s], 0.5f, st1[s]);   // charge (tau=2)
                bool  sp = Ht >= 0.5f;                        // fire (vth=0.5)
                st1[s]   = sp ? 0.0f : Ht;                    // hard reset
                ch.h[j]  = sp ? (_Float16)1.0f : (_Float16)0.0f;
            }
            *(float4*)&s1LDS[(vgrp * 4 + i) * 256 + hgrp * 8] = ch.f4;
        }
        __syncthreads();  // spike tile visible to all waves

        // ---- WMMA: D[g,v] = W2[g,:] @ s1[:,v], K=256 in 8 steps of 32 ----
        const _Float16* pB = &s1LDS[vB * 256 + kb];
        v16h bf[8];
#pragma unroll
        for (int kk = 0; kk < 8; ++kk) {
            const _Float16* p = pB + kk * 32;
            *(float4*)&bf[kk]       = *(const float4*)p;
            *((float4*)&bf[kk] + 1) = *(const float4*)(p + 16);
        }
        v8f acc = {0.f, 0.f, 0.f, 0.f, 0.f, 0.f, 0.f, 0.f};
#pragma unroll
        for (int kk = 0; kk < 8; ++kk) {
            acc = __builtin_amdgcn_wmma_f32_16x16x32_f16(
                      false, af[kk], false, bf[kk], (short)0, acc, false, false);
        }
        // Software pipeline: stay 3 B-fragments (6 ds_load_b128) ahead of the
        // tensor pipe so waits relax from dscnt==0 to a partial drain.
        SGB(0x100, 6, 0);                       // frag 0,1,2 loads
        SGB(0x008, 1, 0); SGB(0x100, 2, 0);     // wmma0 | frag3
        SGB(0x008, 1, 0); SGB(0x100, 2, 0);     // wmma1 | frag4
        SGB(0x008, 1, 0); SGB(0x100, 2, 0);     // wmma2 | frag5
        SGB(0x008, 1, 0); SGB(0x100, 2, 0);     // wmma3 | frag6
        SGB(0x008, 1, 0); SGB(0x100, 2, 0);     // wmma4 | frag7
        SGB(0x008, 3, 0);                       // wmma5,6,7 drain

        // ---- epilogue: BN2 + LIF2 + accumulate spike mean ----------------
#pragma unroll
        for (int r = 0; r < 8; ++r) {
            float h2 = fmaf(acc[r], inv2r[r], bias2r[r]);
            float Ht = fmaf(h2 - st2[r], 0.5f, st2[r]);
            bool  sp = Ht >= 0.5f;
            st2[r]   = sp ? 0.0f : Ht;
            ssum[r] += sp ? 1.0f : 0.0f;
        }
    }

    // Reduce over the 16 N-lanes of each half, then add into feat[b][g].
#pragma unroll
    for (int r = 0; r < 8; ++r) {
        float s = ssum[r];
        s += __shfl_xor(s, 1, 32);
        s += __shfl_xor(s, 2, 32);
        s += __shfl_xor(s, 4, 32);
        s += __shfl_xor(s, 8, 32);
        if ((lane & 15) == 0) {
            int gr = gbase + m * 16 + r + ((lane >> 4) << 3);
            atomicAdd(&feat[b * 256 + gr], s);   // exactly 2 adds -> commutative
        }
    }
}

// Classifier: out[b,nc] = mean_feat[b,:] @ wc[nc,:] + bc[nc].  One wave per b.
__global__ __launch_bounds__(32) void snn_cls(
    const float* __restrict__ feat, const float* __restrict__ wc,
    const float* __restrict__ bc, float* __restrict__ out)
{
    int b = blockIdx.x, lane = threadIdx.x;
    float p[12];
#pragma unroll
    for (int nc = 0; nc < 12; ++nc) p[nc] = 0.0f;
    for (int g = lane; g < 256; g += 32) {
        float f = feat[b * 256 + g] * (1.0f / 2048.0f);  // mean over T=64, V=32
#pragma unroll
        for (int nc = 0; nc < 12; ++nc) p[nc] += f * wc[nc * 256 + g];
    }
#pragma unroll
    for (int nc = 0; nc < 12; ++nc) {
        float s = p[nc];
        s += __shfl_xor(s, 16, 32);
        s += __shfl_xor(s,  8, 32);
        s += __shfl_xor(s,  4, 32);
        s += __shfl_xor(s,  2, 32);
        s += __shfl_xor(s,  1, 32);
        if (lane == 0) out[b * 12 + nc] = s + bc[nc];
    }
}

extern "C" void kernel_launch(void* const* d_in, const int* in_sizes, int n_in,
                              void* d_out, int out_size, void* d_ws, size_t ws_size,
                              hipStream_t stream) {
    const float* x   = (const float*)d_in[0];
    const float* w1  = (const float*)d_in[1];
    const float* b1  = (const float*)d_in[2];
    const float* g1  = (const float*)d_in[3];
    const float* be1 = (const float*)d_in[4];
    const float* m1  = (const float*)d_in[5];
    const float* v1  = (const float*)d_in[6];
    const float* w2  = (const float*)d_in[7];
    const float* b2  = (const float*)d_in[8];
    const float* g2  = (const float*)d_in[9];
    const float* be2 = (const float*)d_in[10];
    const float* m2  = (const float*)d_in[11];
    const float* v2  = (const float*)d_in[12];
    const float* wc  = (const float*)d_in[13];
    const float* bc  = (const float*)d_in[14];

    char* ws = (char*)d_ws;
    _Float16* w2h      = (_Float16*)(ws + WS_W2H);
    float4*   we1pack  = (float4*)  (ws + WS_WE1);
    float2*   invbias2 = (float2*)  (ws + WS_INVB2);
    float*    feat     = (float*)   (ws + WS_FEAT);

    snn_prep<<<256, 256, 0, stream>>>(w1, b1, g1, be1, m1, v1,
                                      w2, b2, g2, be2, m2, v2,
                                      w2h, we1pack, invbias2, feat);
    snn_main<<<dim3(4, 64), 256, 0, stream>>>(x, w2h, we1pack, invbias2, feat);
    snn_cls<<<64, 32, 0, stream>>>(feat, wc, bc, (float*)d_out);
}
